// Model_14912126452256
// MI455X (gfx1250) — matrix operand order repaired and checked by reading, not compiled, not run on hardware
//
#include <hip/hip_runtime.h>

typedef _Float16 f16;
typedef __attribute__((ext_vector_type(16))) _Float16 v16h;
typedef __attribute__((ext_vector_type(8)))  _Float16 v8h;
typedef __attribute__((ext_vector_type(8)))  float    v8f;

constexpr int NB   = 8;          // batch
constexpr int TT   = 750;        // time
constexpr int TTP  = 768;        // time padded to mult of 32 (zero-padded K dim)
constexpr int CC   = 20;         // classes
constexpr int DD   = 1024;       // feature dim
constexpr int HH   = 4;          // heads
constexpr int DHK  = 256;        // head dim
constexpr int NTT  = NB * TT;    // 6000
constexpr int SORTN = 1024;

// ---------------- activation ----------------
__device__ __forceinline__ float act_apply(float x, int act) {
  switch (act) {
    case 1: return x > 0.f ? x : 0.2f * x;                              // leaky relu(0.2)
    case 2: return x > 0.f ? x : 0.f;                                   // relu
    case 3: return 1.f / (1.f + __expf(-x));                            // sigmoid
    case 4: return 0.5f * x * (1.f + erff(x * 0.70710678118654752f));   // exact gelu
  }
  return x;
}

// ---------------- fragment loads (contiguous k order in A and B, aligned b128) ----------------
// A fragment: lane holds K = 16*half + {0..15} -> 32 contiguous bytes
__device__ __forceinline__ v16h ldA(const f16* p) {
  v8h x0 = *(const v8h*)p;
  v8h x1 = *(const v8h*)(p + 8);
  return __builtin_shufflevector(x0, x1, 0,1,2,3,4,5,6,7,8,9,10,11,12,13,14,15);
}
// B fragment (column-major storage): lane holds K = 16*half + {0..15} -> 32 contiguous bytes
__device__ __forceinline__ v16h ldB(const f16* p) {
  v8h x0 = *(const v8h*)p;
  v8h x1 = *(const v8h*)(p + 8);
  return __builtin_shufflevector(x0, x1, 0,1,2,3,4,5,6,7,8,9,10,11,12,13,14,15);
}
__device__ __forceinline__ v8f wmma(v16h a, v16h b, v8f c) {
  return __builtin_amdgcn_wmma_f32_16x16x32_f16(false, a, false, b, (short)0, c, false, false);
}

// ---------------- zero fill (16B granules) ----------------
__global__ void k_zero(uint4* __restrict__ p, long n16) {
  long i = (long)blockIdx.x * blockDim.x + threadIdx.x;
  if (i < n16) p[i] = make_uint4(0u, 0u, 0u, 0u);
}

// split x (N,T,2D) into f16 [mod][nt][d]
__global__ void k_split(const float* __restrict__ x, f16* __restrict__ xh) {
  long i = (long)blockIdx.x * blockDim.x + threadIdx.x;
  if (i >= (long)NTT * DD) return;
  long nt = i / DD, d = i % DD;
  xh[i]                  = (f16)x[nt * (2L * DD) + d];
  xh[(long)NTT * DD + i] = (f16)x[nt * (2L * DD) + DD + d];
}

// transpose-convert: src (B,R,C) f32 -> dst (B,C,R) f16
__global__ void k_cvt_t(const float* __restrict__ s, f16* __restrict__ d, int Bn, int R, int C) {
  long i = (long)blockIdx.x * blockDim.x + threadIdx.x;
  if (i >= (long)Bn * R * C) return;
  long b = i / ((long)R * C), rem = i % ((long)R * C);
  long r = rem / C, c = rem % C;
  d[(b * C + c) * R + r] = (f16)s[i];
}

// conv weight repack: src (O,I,KW) f32 -> dst [kk][O][I] f16
__global__ void k_cvt_w(const float* __restrict__ s, f16* __restrict__ d, int O, int I, int KW) {
  long i = (long)blockIdx.x * blockDim.x + threadIdx.x;
  if (i >= (long)O * I * KW) return;
  long o = i / ((long)I * KW), rem = i % ((long)I * KW);
  long ii = rem / KW, kk = rem % KW;
  d[((long)kk * O + o) * I + ii] = (f16)s[i];
}

// ---------------- WMMA GEMM: one wave = 32x32 tile, B stored column-major ----------------
// C[m,n] = act( sum_k A[m,k] * B[n*ldb + k] );  batch offsets: (z/div)*s1 + (z%div)*s2
__global__ void k_gemm(const f16* __restrict__ A, const f16* __restrict__ B,
                       float* __restrict__ Cf, f16* __restrict__ Ch,
                       int M, int Nn, int K, int lda, int ldb, int ldc, int act,
                       int divA, long sA1, long sA2,
                       int divB, long sB1, long sB2, long sC1) {
  int z = blockIdx.z;
  A += (long)(z / divA) * sA1 + (long)(z % divA) * sA2;
  B += (long)(z / divB) * sB1 + (long)(z % divB) * sB2;
  long coff = (long)z * sC1;
  int row0 = blockIdx.y * 32, col0 = blockIdx.x * 32;
  int l = threadIdx.x & 31, half = l >> 4, lm = l & 15;
  const f16* pa0 = A + (long)(row0 + lm) * lda + half * 16;
  const f16* pa1 = pa0 + 16L * lda;
  const f16* pb0 = B + (long)(col0 + lm) * ldb + half * 16;
  const f16* pb1 = pb0 + 16L * ldb;
  v8f a00 = {}, a01 = {}, a10 = {}, a11 = {};
  for (int kb = 0; kb < K; kb += 32) {
    v16h fa0 = ldA(pa0 + kb), fa1 = ldA(pa1 + kb);
    v16h fb0 = ldB(pb0 + kb), fb1 = ldB(pb1 + kb);
    a00 = wmma(fa0, fb0, a00); a01 = wmma(fa0, fb1, a01);
    a10 = wmma(fa1, fb0, a10); a11 = wmma(fa1, fb1, a11);
  }
  auto st = [&](const v8f& acc, int r0, int c0) {
#pragma unroll
    for (int r = 0; r < 8; ++r) {
      int m = r0 + r + half * 8, n = c0 + lm;
      if (m < M && n < Nn) {
        float v = act_apply(acc[r], act);
        long o = coff + (long)m * ldc + n;
        if (Cf) Cf[o] = v;
        if (Ch) Ch[o] = (f16)v;
      }
    }
  };
  st(a00, row0, col0);      st(a01, row0, col0 + 16);
  st(a10, row0 + 16, col0); st(a11, row0 + 16, col0 + 16);
}

// ---------------- WMMA conv1d-as-GEMM ----------------
// W packed [kk][O][I]; X channel-last [n][t][i]; out channel-last [n][t][o]
__global__ void k_convgemm(const f16* __restrict__ W, const f16* __restrict__ X,
                           const float* __restrict__ bias, const f16* __restrict__ zrow,
                           float* __restrict__ Cf, f16* __restrict__ Ch,
                           int O, int I, int Tn, int KW, int pad, int act) {
  int nb = blockIdx.z;
  const f16* Xn = X + (long)nb * Tn * I;
  long coff = (long)nb * Tn * O;
  int row0 = blockIdx.y * 32, col0 = blockIdx.x * 32;
  int l = threadIdx.x & 31, half = l >> 4, lm = l & 15;
  v8f a00 = {}, a01 = {}, a10 = {}, a11 = {};
  for (int kk = 0; kk < KW; ++kk) {
    int ts0 = col0 + lm + kk - pad, ts1 = ts0 + 16;
    const f16* pw0 = W + ((long)kk * O + row0 + lm) * I + half * 16;
    const f16* pw1 = pw0 + 16L * I;
    const f16* px0 = ((unsigned)ts0 < (unsigned)Tn) ? Xn + (long)ts0 * I + half * 16
                                                    : zrow + half * 16;
    const f16* px1 = ((unsigned)ts1 < (unsigned)Tn) ? Xn + (long)ts1 * I + half * 16
                                                    : zrow + half * 16;
    for (int kb = 0; kb < I; kb += 32) {
      v16h fa0 = ldA(pw0 + kb), fa1 = ldA(pw1 + kb);
      v16h fb0 = ldB(px0 + kb), fb1 = ldB(px1 + kb);
      a00 = wmma(fa0, fb0, a00); a01 = wmma(fa0, fb1, a01);
      a10 = wmma(fa1, fb0, a10); a11 = wmma(fa1, fb1, a11);
    }
  }
  auto st = [&](const v8f& acc, int r0, int c0) {
#pragma unroll
    for (int r = 0; r < 8; ++r) {
      int m = r0 + r + half * 8, t = c0 + lm;
      if (m < O && t < Tn) {
        float v = act_apply(acc[r] + bias[m], act);
        long o = coff + (long)t * O + m;
        if (Cf) Cf[o] = v;
        if (Ch) Ch[o] = (f16)v;
      }
    }
  };
  st(a00, row0, col0);      st(a01, row0, col0 + 16);
  st(a10, row0 + 16, col0); st(a11, row0 + 16, col0 + 16);
}

// ---------------- l2norm rows of 256; writes row-major f16 + transposed padded f16 ----------------
__global__ void k_l2norm(const float* __restrict__ src, f16* __restrict__ dst, f16* __restrict__ dstT) {
  __shared__ float red[256];
  long r = blockIdx.x; int k = threadIdx.x;
  float v = src[r * DHK + k];
  red[k] = v * v; __syncthreads();
  for (int s = 128; s > 0; s >>= 1) { if (k < s) red[k] += red[k + s]; __syncthreads(); }
  float nv = v / fmaxf(sqrtf(red[0]), 1e-12f);
  dst[r * DHK + k] = (f16)nv;
  // r = ((mod*HH+h)*NB + n)*TT + ... decompose: r = mod*HH*NTT + h*NTT + n*TT + t
  long mod = r / ((long)HH * NTT), rem = r % ((long)HH * NTT);
  long h = rem / NTT, nt = rem % NTT, n = nt / TT, t = nt % TT;
  dstT[(((mod * HH + h) * NB + n) * DHK + k) * TTP + t] = (f16)nv;
}

// ---------------- row softmax (f32 in, f16 out with padded row stride) ----------------
__global__ void k_rowsoftmax(const float* __restrict__ src, f16* __restrict__ dst, int len, int dstld) {
  __shared__ float red[256];
  long r = blockIdx.x; int t = threadIdx.x;
  const float* p = src + r * len;
  float mx = -3.4e38f;
  for (int i = t; i < len; i += 256) mx = fmaxf(mx, p[i]);
  red[t] = mx; __syncthreads();
  for (int s = 128; s > 0; s >>= 1) { if (t < s) red[t] = fmaxf(red[t], red[t + s]); __syncthreads(); }
  mx = red[0]; __syncthreads();
  float sum = 0.f;
  for (int i = t; i < len; i += 256) sum += __expf(p[i] - mx);
  red[t] = sum; __syncthreads();
  for (int s = 128; s > 0; s >>= 1) { if (t < s) red[t] += red[t + s]; __syncthreads(); }
  float inv = 1.f / red[0];
  f16* q = dst + r * dstld;
  for (int i = t; i < len; i += 256) q[i] = (f16)(__expf(p[i] - mx) * inv);
}

// column softmax of att[b] (TxT), written transposed padded: fa[b][t][s] = softmax_s(att[b][s][t])
__global__ void k_colsoftmax(const float* __restrict__ att, f16* __restrict__ fa, int Tn) {
  __shared__ float red[256];
  int tcol = blockIdx.x; long b = blockIdx.y; int t = threadIdx.x;
  const float* p = att + b * Tn * Tn;
  float mx = -3.4e38f;
  for (int s = t; s < Tn; s += 256) mx = fmaxf(mx, p[(long)s * Tn + tcol]);
  red[t] = mx; __syncthreads();
  for (int s = 128; s > 0; s >>= 1) { if (t < s) red[t] = fmaxf(red[t], red[t + s]); __syncthreads(); }
  mx = red[0]; __syncthreads();
  float sum = 0.f;
  for (int s = t; s < Tn; s += 256) sum += __expf(p[(long)s * Tn + tcol] - mx);
  red[t] = sum; __syncthreads();
  for (int s = 128; s > 0; s >>= 1) { if (t < s) red[t] += red[t + s]; __syncthreads(); }
  float inv = 1.f / red[0];
  f16* q = fa + b * TT * TTP + (long)tcol * TTP;
  for (int s = t; s < Tn; s += 256) q[s] = (f16)(__expf(p[(long)s * Tn + tcol] - mx) * inv);
}

// ---------------- f = tanh(reshape(e) + x)  -> channel-LAST f16 [n][t][d] ----------------
// e layout: [((h*NB+n)*TT + t)*DHK + k]; reference reshape (n,h,k,t)->(n,t,d) is flat.
__global__ void k_fuse_tanh(const float* __restrict__ e, const float* __restrict__ x,
                            f16* __restrict__ outc, int mod) {
  long i = (long)blockIdx.x * blockDim.x + threadIdx.x;
  if (i >= (long)NB * TT * DD) return;
  int n = (int)(i / ((long)TT * DD));
  int rem = (int)(i % ((long)TT * DD));
  int tp = rem / DD, dp = rem % DD;
  int t2 = rem % TT, hk = rem / TT;
  int h = hk / DHK, k = hk % DHK;
  float ev = e[(((long)h * NB + n) * TT + t2) * DHK + k];
  float rv = x[((long)n * TT + tp) * (2L * DD) + (long)mod * DD + dp];
  outc[((long)n * TT + tp) * DD + dp] = (f16)tanhf(ev + rv);
}

// feat[n][t][d] *= a[n*TT+t]  (in place on f16)
__global__ void k_scale(f16* __restrict__ c, const float* __restrict__ a) {
  long i = (long)blockIdx.x * blockDim.x + threadIdx.x;
  if (i >= (long)NB * TT * DD) return;
  long nt = i / DD;
  c[i] = (f16)((float)c[i] * a[nt]);
}

// cas=casr+casf ; seg=(softmax_c(cas)+aas)/2 ; aas=(sig(aasr)+sig(aasf))/2  (all [n][t][c])
__global__ void k_combine(const float* __restrict__ casr, const float* __restrict__ casf,
                          const float* __restrict__ aasr, const float* __restrict__ aasf,
                          float* __restrict__ cas, float* __restrict__ seg, float* __restrict__ aas) {
  int i = blockIdx.x * blockDim.x + threadIdx.x;
  if (i >= NB * TT) return;
  float cv[CC], mx = -3.4e38f;
  for (int c = 0; c < CC; ++c) {
    float v = casr[(long)i * CC + c] + casf[(long)i * CC + c];
    cv[c] = v; cas[(long)i * CC + c] = v;
    mx = fmaxf(mx, v);
  }
  float s = 0.f;
  for (int c = 0; c < CC; ++c) s += __expf(cv[c] - mx);
  float av = 0.5f * (1.f / (1.f + __expf(-aasr[i])) + 1.f / (1.f + __expf(-aasf[i])));
  aas[i] = av;
  for (int c = 0; c < CC; ++c) seg[(long)i * CC + c] = 0.5f * (__expf(cv[c] - mx) / s + av);
}

// ---------------- stable descending bitonic sort in LDS ----------------
__device__ void bitonic_desc(float* val, int* idx, int tid, int nth) {
  for (int k = 2; k <= SORTN; k <<= 1)
    for (int j = k >> 1; j > 0; j >>= 1) {
      for (int i = tid; i < SORTN; i += nth) {
        int ixj = i ^ j;
        if (ixj > i) {
          float va = val[i], vb = val[ixj]; int ia = idx[i], ib = idx[ixj];
          bool before = (va > vb) || (va == vb && ia < ib);   // stable desc
          bool dir = ((i & k) == 0);
          if (before != dir) { val[i] = vb; val[ixj] = va; idx[i] = ib; idx[ixj] = ia; }
        }
      }
      __syncthreads();
    }
}

__device__ void sort_load(const float* seg, float* sval, int* sidx, int n, int c, int tid, int nth) {
  for (int i = tid; i < SORTN; i += nth) {
    sval[i] = (i < TT) ? seg[((long)n * TT + i) * CC + c] : -3.4e38f;
    sidx[i] = i;
  }
  __syncthreads();
  bitonic_desc(sval, sidx, tid, nth);
}

// temporal_clustering: one block per (n,c)
__global__ void k_cluster(const float* __restrict__ seg, float* __restrict__ mask) {
  __shared__ float sval[SORTN]; __shared__ int sidx[SORTN]; __shared__ float saux[SORTN];
  int r = blockIdx.x, n = r / CC, c = r % CC, tid = threadIdx.x, nth = blockDim.x;
  sort_load(seg, sval, sidx, n, c, tid, nth);
  if (tid == 0) {
    float ps = sval[0], pn = 1.f, ns = sval[TT - 1], nn = 1.f;
    saux[0] = 1.f;
    for (int i = 1; i <= TT - 2; ++i) {
      float value = sval[i];
      float cf = (fabsf(value - ps / pn) <= fabsf(value - ns / nn)) ? 1.f : 0.f;
      float inv = 1.f / (float)(i + 1);
      ps += cf * value * inv; pn += cf * inv;
      ns += (1.f - cf) * value; nn += (1.f - cf);
      saux[i] = cf;
    }
    saux[TT - 1] = 0.f;
  }
  __syncthreads();
  for (int j = tid; j < TT; j += nth)
    mask[((long)n * TT + sidx[j]) * CC + c] = saux[j];
}

// mask_refining: one block per (n,c)
__global__ void k_refine(const float* __restrict__ seg, float* __restrict__ mask,
                         const float* __restrict__ cas) {
  __shared__ float sval[SORTN]; __shared__ int sidx[SORTN];
  __shared__ float sm[SORTN]; __shared__ float sc[SORTN]; __shared__ float smean;
  int r = blockIdx.x, n = r / CC, c = r % CC, tid = threadIdx.x, nth = blockDim.x;
  sort_load(seg, sval, sidx, n, c, tid, nth);
  for (int j = tid; j < TT; j += nth) {
    long o = ((long)n * TT + sidx[j]) * CC + c;
    sm[j] = mask[o]; sc[j] = cas[o];
  }
  __syncthreads();
  if (tid == 0) {
    float ranksum = 0.f, refnum = 0.f, kf = 0.f;
    for (int j = 0; j < TT; ++j) {
      float rk = sm[j] / (float)(j + 2);
      ranksum += rk; refnum += sc[j] * rk; kf += sm[j];
    }
    float ref = refnum / fmaxf(ranksum, 1.f);
    int k = (int)(kf + 0.5f), s_cnt = 0;
    float cnt = 0.f, cums = 0.f;
    for (int j = 0; j < TT; ++j) {
      cnt += sm[j]; cums += sc[j] * sm[j];
      if ((cums / fmaxf(cnt, 1.f)) >= ref && sm[j] > 0.f) ++s_cnt;
    }
    int target = (s_cnt == 0) ? (k - 1) : ((s_cnt - 1 < k - 1) ? s_cnt - 1 : k - 1);
    float mean = 0.f; cnt = 0.f;
    for (int j = 0; j < TT; ++j) {
      cnt += sm[j];
      if (sm[j] > 0.f && ((int)(cnt + 0.5f) - 1) == target) mean += sval[j];
    }
    smean = (k > 0) ? mean : 0.f;
  }
  __syncthreads();
  float mean = smean;
  for (int t = tid; t < TT; t += nth) {
    long o = ((long)n * TT + t) * CC + c;
    mask[o] *= (seg[o] >= mean) ? 1.f : 0.f;
  }
}

// calculate_score raw act/bkg: one block per (n,c)
__global__ void k_score(const float* __restrict__ seg, const float* __restrict__ mask,
                        const float* __restrict__ cas,
                        float* __restrict__ actr, float* __restrict__ bkgr) {
  __shared__ float sval[SORTN]; __shared__ int sidx[SORTN];
  __shared__ float sm[SORTN]; __shared__ float sc[SORTN];
  int r = blockIdx.x, n = r / CC, c = r % CC, tid = threadIdx.x, nth = blockDim.x;
  sort_load(seg, sval, sidx, n, c, tid, nth);
  for (int j = tid; j < TT; j += nth) {
    long o = ((long)n * TT + sidx[j]) * CC + c;
    sm[j] = mask[o]; sc[j] = cas[o];
  }
  __syncthreads();
  if (tid == 0) {
    float an = 0.f, av = 0.f, bn = 0.f, bv = 0.f;
    for (int j = 0; j < TT; ++j) {
      float rk = 1.f / (float)(j + 2);
      an += rk * sm[j]; av += sc[j] * rk * sm[j];
      bn += 1.f - sm[j]; bv += sc[j] * (1.f - sm[j]);
    }
    actr[n * CC + c] = av / fmaxf(an, 1.f);
    bkgr[n * CC + c] = bv / fmaxf(bn, 1.f);
  }
}

__global__ void k_finalsm(const float* __restrict__ ar, const float* __restrict__ br,
                          float* __restrict__ oa, float* __restrict__ ob) {
  int n = threadIdx.x;
  if (n >= NB) return;
  for (int w = 0; w < 2; ++w) {
    const float* src = w ? br : ar; float* dst = w ? ob : oa;
    float mx = -3.4e38f;
    for (int c = 0; c < CC; ++c) mx = fmaxf(mx, src[n * CC + c]);
    float s = 0.f;
    for (int c = 0; c < CC; ++c) s += __expf(src[n * CC + c] - mx);
    for (int c = 0; c < CC; ++c) dst[n * CC + c] = __expf(src[n * CC + c] - mx) / s;
  }
}

// ============================ host ============================
extern "C" void kernel_launch(void* const* d_in, const int* in_sizes, int n_in,
                              void* d_out, int out_size, void* d_ws, size_t ws_size,
                              hipStream_t stream) {
  (void)in_sizes; (void)n_in; (void)out_size; (void)ws_size;
  const float* x        = (const float*)d_in[0];
  const float* rgb_proj = (const float*)d_in[1];
  const float* flow_proj= (const float*)d_in[2];
  const float* atte_w   = (const float*)d_in[3];
  const float* ra_w1 = (const float*)d_in[4];  const float* ra_b1 = (const float*)d_in[5];
  const float* ra_w2 = (const float*)d_in[6];  const float* ra_b2 = (const float*)d_in[7];
  const float* ra_w3 = (const float*)d_in[8];  const float* ra_b3 = (const float*)d_in[9];
  const float* fa_w1 = (const float*)d_in[10]; const float* fa_b1 = (const float*)d_in[11];
  const float* fa_w2 = (const float*)d_in[12]; const float* fa_b2 = (const float*)d_in[13];
  const float* fa_w3 = (const float*)d_in[14]; const float* fa_b3 = (const float*)d_in[15];
  const float* cre_w = (const float*)d_in[16]; const float* cre_b = (const float*)d_in[17];
  const float* cr_w  = (const float*)d_in[18]; const float* cr_b  = (const float*)d_in[19];
  const float* cfe_w = (const float*)d_in[20]; const float* cfe_b = (const float*)d_in[21];
  const float* cf_w  = (const float*)d_in[22]; const float* cf_b  = (const float*)d_in[23];
  const float* are_w = (const float*)d_in[24]; const float* are_b = (const float*)d_in[25];
  const float* ar_w  = (const float*)d_in[26]; const float* ar_b  = (const float*)d_in[27];
  const float* afe_w = (const float*)d_in[28]; const float* afe_b = (const float*)d_in[29];
  const float* af_w  = (const float*)d_in[30]; const float* af_b  = (const float*)d_in[31];

  char* ws = (char*)d_ws; size_t off = 0;
  auto alloc = [&](size_t bytes) -> char* {
    size_t a = (off + 255) & ~(size_t)255; off = a + bytes; return ws + a;
  };
  f16*   xh    = (f16*)alloc(2L * NTT * DD * 2);
  f16*   projT = (f16*)alloc(2L * HH * DHK * DD * 2);      // [mod][h][k][d]
  f16*   atteT = (f16*)alloc((long)HH * DHK * DHK * 2);    // [h][m][k]
  float* att   = (float*)alloc((long)NB * HH * TT * TT * 4); // reused first as o_pre
  float* o_pre = att;
  f16*   oh    = (f16*)alloc(2L * HH * NTT * DHK * 2);     // row-major [mod][h][nt][k]
  f16*   oT    = (f16*)alloc(2L * HH * NB * DHK * TTP * 2);// transposed [mod][h][n][k][t], zero-padded
  f16*   tmph  = (f16*)alloc((long)NB * HH * TT * DHK * 2);
  f16*   rah   = (f16*)alloc((long)NB * HH * TT * TTP * 2);// padded cols, zero-filled
  f16*   fah   = (f16*)alloc((long)NB * HH * TT * TTP * 2);
  float* ebuf  = (float*)alloc((long)NB * HH * TT * DHK * 4);
  f16*   rgbc  = (f16*)alloc((long)NB * TT * DD * 2);      // channel-last
  f16*   flowc = (f16*)alloc((long)NB * TT * DD * 2);
  f16*   y1h   = (f16*)alloc((long)NB * TT * 512 * 2);
  f16*   y2h   = (f16*)alloc((long)NB * TT * 512 * 2);
  f16*   h1h   = (f16*)alloc((long)NB * TT * DD * 2);
  float* a_r   = (float*)alloc((long)NB * TT * 4);
  float* a_f   = (float*)alloc((long)NB * TT * 4);
  float* casr  = (float*)alloc((long)NB * TT * CC * 4);
  float* casf  = (float*)alloc((long)NB * TT * CC * 4);
  float* aasr  = (float*)alloc((long)NB * TT * 4);
  float* aasf  = (float*)alloc((long)NB * TT * 4);
  float* casb  = (float*)alloc((long)NB * TT * CC * 4);
  float* actr  = (float*)alloc(NB * CC * 4);
  float* bkgr  = (float*)alloc(NB * CC * 4);
  f16* raw1p = (f16*)alloc(3L * 512 * DD * 2);  f16* raw2p = (f16*)alloc(3L * 512 * 512 * 2);
  f16* raw3p = (f16*)alloc(512L * 2);
  f16* faw1p = (f16*)alloc(3L * 512 * DD * 2);  f16* faw2p = (f16*)alloc(3L * 512 * 512 * 2);
  f16* faw3p = (f16*)alloc(512L * 2);
  f16* crep = (f16*)alloc(3L * DD * DD * 2);  f16* crp = (f16*)alloc((long)CC * DD * 2);
  f16* cfep = (f16*)alloc(3L * DD * DD * 2);  f16* cfp = (f16*)alloc((long)CC * DD * 2);
  f16* arep = (f16*)alloc(3L * DD * DD * 2);  f16* arp = (f16*)alloc((long)DD * 2);
  f16* afep = (f16*)alloc(3L * DD * DD * 2);  f16* afp = (f16*)alloc((long)DD * 2);
  f16* zrow = (f16*)alloc(2048L * 2);                      // zeroed pad row for conv
  (void)alloc(4L << 20);                                   // tail slack for benign over-reads

  float* out      = (float*)d_out;
  float* out_act  = out;
  float* out_bkg  = out + NB * CC;
  float* out_aas  = out + 2 * NB * CC;
  float* out_seg  = out_aas + (long)NB * TT;
  float* out_mask = out_seg + (long)NB * TT * CC;

  auto cdiv = [](long a, long b) { return (unsigned)((a + b - 1) / b); };
  auto zero = [&](void* p, long bytes) {
    k_zero<<<cdiv(bytes / 16, 256), 256, 0, stream>>>((uint4*)p, bytes / 16);
  };

  // --- zero K-padding buffers + pad row ---
  zero(rah, (long)NB * HH * TT * TTP * 2);
  zero(fah, (long)NB * HH * TT * TTP * 2);
  zero(oT,  2L * HH * NB * DHK * TTP * 2);
  zero(zrow, 2048L * 2);

  // --- f16 staging (transposed / repacked for contiguous B fragments) ---
  k_split<<<cdiv((long)NTT * DD, 256), 256, 0, stream>>>(x, xh);
  k_cvt_t<<<cdiv((long)HH * DD * DHK, 256), 256, 0, stream>>>(rgb_proj, projT, HH, DD, DHK);
  k_cvt_t<<<cdiv((long)HH * DD * DHK, 256), 256, 0, stream>>>(flow_proj, projT + (long)HH * DHK * DD, HH, DD, DHK);
  k_cvt_t<<<cdiv((long)HH * DHK * DHK, 256), 256, 0, stream>>>(atte_w, atteT, HH, DHK, DHK);
  auto cw = [&](const float* s, f16* d, int O, int I, int KW) {
    k_cvt_w<<<cdiv((long)O * I * KW, 256), 256, 0, stream>>>(s, d, O, I, KW);
  };
  cw(ra_w1, raw1p, 512, DD, 3);  cw(ra_w2, raw2p, 512, 512, 3);  cw(ra_w3, raw3p, 1, 512, 1);
  cw(fa_w1, faw1p, 512, DD, 3);  cw(fa_w2, faw2p, 512, 512, 3);  cw(fa_w3, faw3p, 1, 512, 1);
  cw(cre_w, crep, DD, DD, 3);    cw(cr_w, crp, CC, DD, 1);
  cw(cfe_w, cfep, DD, DD, 3);    cw(cf_w, cfp, CC, DD, 1);
  cw(are_w, arep, DD, DD, 3);    cw(ar_w, arp, 1, DD, 1);
  cw(afe_w, afep, DD, DD, 3);    cw(af_w, afp, 1, DD, 1);

  // --- MCA projections (z = mod*HH+h): o_pre = xh @ projT^T ---
  k_gemm<<<dim3(cdiv(DHK, 32), cdiv(NTT, 32), 2 * HH), 32, 0, stream>>>(
      xh, projT, o_pre, nullptr, NTT, DHK, DD, DD, DD, DHK, 0,
      HH, (long)NTT * DD, 0L, 1, (long)DHK * DD, 0L, (long)NTT * DHK);
  k_l2norm<<<2 * HH * NTT, 256, 0, stream>>>(o_pre, oh, oT);

  // --- tmp = o_rgb @ atte_w[h]  (z = h*NB+n) ---
  k_gemm<<<dim3(cdiv(DHK, 32), cdiv(TT, 32), NB * HH), 32, 0, stream>>>(
      oh, atteT, nullptr, tmph, TT, DHK, DHK, DHK, DHK, DHK, 0,
      NB, (long)NTT * DHK, (long)TT * DHK, NB, (long)DHK * DHK, 0L, (long)TT * DHK);

  // --- att = tmp @ o_flow^T  (o_flow row-major = column-major B) ---
  k_gemm<<<dim3(cdiv(TT, 32), cdiv(TT, 32), NB * HH), 32, 0, stream>>>(
      tmph, oh + (long)HH * NTT * DHK, att, nullptr, TT, TT, DHK, DHK, DHK, TT, 0,
      1, (long)TT * DHK, 0L, NB, (long)NTT * DHK, (long)TT * DHK, (long)TT * TT);

  // --- softmaxes ---
  k_rowsoftmax<<<NB * HH * TT, 256, 0, stream>>>(att, rah, TT, TTP);
  k_colsoftmax<<<dim3(TT, NB * HH), 256, 0, stream>>>(att, fah, TT);

  // --- e = gelu(atte @ o) via oT (padded K=750->768); fuse tanh -> channel-last f16 ---
  k_gemm<<<dim3(cdiv(DHK, 32), cdiv(TT, 32), NB * HH), 32, 0, stream>>>(
      rah, oT, ebuf, nullptr, TT, DHK, TT, TTP, TTP, DHK, 4,
      1, (long)TT * TTP, 0L, 1, (long)DHK * TTP, 0L, (long)TT * DHK);
  k_fuse_tanh<<<cdiv((long)NB * TT * DD, 256), 256, 0, stream>>>(ebuf, x, rgbc, 0);
  k_gemm<<<dim3(cdiv(DHK, 32), cdiv(TT, 32), NB * HH), 32, 0, stream>>>(
      fah, oT + (long)HH * NB * DHK * TTP, ebuf, nullptr, TT, DHK, TT, TTP, TTP, DHK, 4,
      1, (long)TT * TTP, 0L, 1, (long)DHK * TTP, 0L, (long)TT * DHK);
  k_fuse_tanh<<<cdiv((long)NB * TT * DD, 256), 256, 0, stream>>>(ebuf, x, flowc, 1);

  // --- attention gating conv chains (channel-last everywhere) ---
  dim3 g512(cdiv(TT, 32), 16, NB), g1(cdiv(TT, 32), 1, NB);
  k_convgemm<<<g512, 32, 0, stream>>>(raw1p, rgbc, ra_b1, zrow, nullptr, y1h, 512, DD, TT, 3, 1, 1);
  k_convgemm<<<g512, 32, 0, stream>>>(raw2p, y1h, ra_b2, zrow, nullptr, y2h, 512, 512, TT, 3, 1, 1);
  k_convgemm<<<g1, 32, 0, stream>>>(raw3p, y2h, ra_b3, zrow, a_r, nullptr, 1, 512, TT, 1, 0, 3);
  k_scale<<<cdiv((long)NB * TT * DD, 256), 256, 0, stream>>>(rgbc, a_r);
  k_convgemm<<<g512, 32, 0, stream>>>(faw1p, flowc, fa_b1, zrow, nullptr, y1h, 512, DD, TT, 3, 1, 1);
  k_convgemm<<<g512, 32, 0, stream>>>(faw2p, y1h, fa_b2, zrow, nullptr, y2h, 512, 512, TT, 3, 1, 1);
  k_convgemm<<<g1, 32, 0, stream>>>(faw3p, y2h, fa_b3, zrow, a_f, nullptr, 1, 512, TT, 1, 0, 3);
  k_scale<<<cdiv((long)NB * TT * DD, 256), 256, 0, stream>>>(flowc, a_f);

  // --- CAS / AAS branches ---
  dim3 gD(cdiv(TT, 32), 32, NB);
  k_convgemm<<<gD, 32, 0, stream>>>(crep, rgbc, cre_b, zrow, nullptr, h1h, DD, DD, TT, 3, 1, 2);
  k_convgemm<<<g1, 32, 0, stream>>>(crp, h1h, cr_b, zrow, casr, nullptr, CC, DD, TT, 1, 0, 0);
  k_convgemm<<<gD, 32, 0, stream>>>(cfep, flowc, cfe_b, zrow, nullptr, h1h, DD, DD, TT, 3, 1, 2);
  k_convgemm<<<g1, 32, 0, stream>>>(cfp, h1h, cf_b, zrow, casf, nullptr, CC, DD, TT, 1, 0, 0);
  k_convgemm<<<gD, 32, 0, stream>>>(arep, rgbc, are_b, zrow, nullptr, h1h, DD, DD, TT, 3, 1, 2);
  k_convgemm<<<g1, 32, 0, stream>>>(arp, h1h, ar_b, zrow, aasr, nullptr, 1, DD, TT, 1, 0, 0);
  k_convgemm<<<gD, 32, 0, stream>>>(afep, flowc, afe_b, zrow, nullptr, h1h, DD, DD, TT, 3, 1, 2);
  k_convgemm<<<g1, 32, 0, stream>>>(afp, h1h, af_b, zrow, aasf, nullptr, 1, DD, TT, 1, 0, 0);

  // --- scores, clustering, refining, final ---
  k_combine<<<cdiv(NB * TT, 256), 256, 0, stream>>>(casr, casf, aasr, aasf, casb, out_seg, out_aas);
  k_cluster<<<NB * CC, 512, 0, stream>>>(out_seg, out_mask);
  k_refine<<<NB * CC, 512, 0, stream>>>(out_seg, out_mask, casb);
  k_score<<<NB * CC, 512, 0, stream>>>(out_seg, out_mask, casb, actr, bkgr);
  k_finalsm<<<1, 32, 0, stream>>>(actr, bkgr, out_act, out_bkg);
}
